// CobraBlock_43860206027249
// MI455X (gfx1250) — matrix operand order repaired and checked by reading, not compiled, not run on hardware
//
#include <hip/hip_runtime.h>
#include <hip/hip_bf16.h>

// ---------------------------------------------------------------------------
// CobraBlock (Mamba-style) for gfx1250 / MI455X.
// GEMMs: bf16 hi/lo split (3x v_wmma_f32_16x16x32_bf16) with fp32 accumulate.
// Big GEMMs: LDS-tiled, double-buffered async global->LDS staging (ASYNCcnt)
// overlapping the WMMA pipeline; one workgroup barrier per K-stage.
// ---------------------------------------------------------------------------

#define BATCH   16
#define SEQ     64
#define DIM     2048
#define DTRANK  128
#define DSTATE  16
#define MROWS   (BATCH * SEQ)   // 1024

typedef __attribute__((ext_vector_type(16))) __bf16 v16bf;
typedef __attribute__((ext_vector_type(8)))  __bf16 v8bf;
typedef __attribute__((ext_vector_type(8)))  float  v8f;

struct FragPair { v16bf hi; v16bf lo; };

// ---- fragment loads (bf16 arrays, global or LDS) --------------------------
__device__ __forceinline__ v16bf ldA16(const __bf16* p, int lda, int lane) {
    const __bf16* r = p + (lane & 15) * lda + ((lane & 16) ? 16 : 0);
    v8bf a = *(const v8bf*)(r);
    v8bf b = *(const v8bf*)(r + 8);
    return __builtin_shufflevector(a, b, 0,1,2,3,4,5,6,7,8,9,10,11,12,13,14,15);
}
__device__ __forceinline__ v16bf ldB16(const __bf16* p, int ldb, int lane) {
    return *(const v16bf*)(p + (lane & 15) * ldb + ((lane & 16) ? 16 : 0));
}
__device__ __forceinline__ FragPair ldA(const __bf16* ph, const __bf16* pl, int lda, int lane) {
    FragPair f; f.hi = ldA16(ph, lda, lane); f.lo = ldA16(pl, lda, lane); return f;
}
__device__ __forceinline__ FragPair ldB(const __bf16* ph, const __bf16* pl, int ldb, int lane) {
    FragPair f; f.hi = ldB16(ph, ldb, lane); f.lo = ldB16(pl, ldb, lane); return f;
}

// hi/lo split product: A*B ~= Ah*Bh + Ah*Bl + Al*Bh  (fp32 accumulate)
__device__ __forceinline__ v8f wmma3(const FragPair& a, const FragPair& b, v8f c) {
    c = __builtin_amdgcn_wmma_f32_16x16x32_bf16(false, a.hi, false, b.hi, (short)0, c, false, false);
    c = __builtin_amdgcn_wmma_f32_16x16x32_bf16(false, a.hi, false, b.lo, (short)0, c, false, false);
    c = __builtin_amdgcn_wmma_f32_16x16x32_bf16(false, a.lo, false, b.hi, (short)0, c, false, false);
    return c;
}

__device__ __forceinline__ float silu_f(float v) { return v / (1.0f + __expf(-v)); }

__device__ __forceinline__ void split_store(float v, __bf16* hp, __bf16* lp, int idx) {
    __bf16 h = (__bf16)v;
    hp[idx] = h;
    lp[idx] = (__bf16)(v - (float)h);
}

// LDS byte address = low 32 bits of the flat pointer (aperture in addr[63:32]).
__device__ __forceinline__ unsigned lds_off(const void* p) {
    return (unsigned)(unsigned long long)p;
}

// Async copy of one 16B chunk: global -> LDS (ASYNCcnt-tracked).
__device__ __forceinline__ void async_cp16(const __bf16* gsrc, __bf16* ldst) {
    asm volatile("global_load_async_to_lds_b128 %0, %1, off"
                 :: "v"(lds_off(ldst)), "v"(gsrc) : "memory");
}
__device__ __forceinline__ void async_wait0() {
    asm volatile("s_wait_asynccnt 0x0" ::: "memory");
}

// ---------------------------------------------------------------------------
// Elementwise bf16 hi/lo splitter (run once per operand).
// ---------------------------------------------------------------------------
__global__ void cobra_split(const float* __restrict__ src,
                            __bf16* __restrict__ hi, __bf16* __restrict__ lo, int n) {
    int i = blockIdx.x * 256 + threadIdx.x;
    if (i < n) split_store(src[i], hi, lo, i);
}

// ---------------------------------------------------------------------------
// LDS-tiled GEMM: C[M,N] = A[M,K]*B[N,K]^T (+bias)(+epilogue).
// Block = 256 thr = 8 waves (2M x 4N), block tile 64x128, K staged 32-deep,
// double-buffered async global->LDS; each wave computes a 32x32 macro-tile.
// mode 0: +bias; 1: softplus(+bias); 2: +bias +skip.
// Requires M%64==0, N%128==0, K%32==0.
// ---------------------------------------------------------------------------
#define LSTR 40   // LDS panel row stride (halfs): 80B, 16B-aligned, bank-spread

__global__ void cobra_gemm_lds(const __bf16* __restrict__ Ah, const __bf16* __restrict__ Al, int lda,
                               const __bf16* __restrict__ Bh, const __bf16* __restrict__ Bl, int ldb,
                               float* __restrict__ C, int ldc,
                               const float* __restrict__ bias,
                               const float* __restrict__ skip,
                               int M, int N, int K, int mode) {
    __shared__ __bf16 sAh[2][64 * LSTR],  sAl[2][64 * LSTR];
    __shared__ __bf16 sBh[2][128 * LSTR], sBl[2][128 * LSTR];

    const int lane = threadIdx.x & 31;
    const int wave = threadIdx.x >> 5;
    const int nb   = N >> 7;                  // N / 128
    const int rowB = (blockIdx.x / nb) << 6;  // * 64
    const int colB = (blockIdx.x % nb) << 7;  // * 128

    const int row0 = (wave >> 2) * 32;        // wave tile inside block tile
    const int col0 = (wave & 3) * 32;

    // per-thread copy slots (6 x b128 per stage)
    const int ra = threadIdx.x >> 2;
    const int oa = (threadIdx.x & 3) * 8;

    auto stage = [&](int k0, int bb) {
        async_cp16(Ah + (rowB + ra) * lda + k0 + oa, &sAh[bb][ra * LSTR + oa]);
        async_cp16(Al + (rowB + ra) * lda + k0 + oa, &sAl[bb][ra * LSTR + oa]);
#pragma unroll
        for (int hlf = 0; hlf < 2; ++hlf) {
            const int c  = hlf * 256 + threadIdx.x;
            const int rb = c >> 2;
            const int ob = (c & 3) * 8;
            async_cp16(Bh + (colB + rb) * ldb + k0 + ob, &sBh[bb][rb * LSTR + ob]);
            async_cp16(Bl + (colB + rb) * ldb + k0 + ob, &sBl[bb][rb * LSTR + ob]);
        }
    };

    v8f z = {0.f,0.f,0.f,0.f,0.f,0.f,0.f,0.f};
    v8f acc00 = z, acc01 = z, acc10 = z, acc11 = z;

    const int nst = K >> 5;
    stage(0, 0);

    for (int s = 0; s < nst; ++s) {
        const int cur = s & 1;
        async_wait0();        // stage-s panels landed (overlapped with stage s-1 math)
        __syncthreads();      // all waves' copies visible; prev buffer fully consumed
        if (s + 1 < nst) stage((s + 1) << 5, cur ^ 1);   // overlaps with math below

        FragPair a0 = ldA(&sAh[cur][row0 * LSTR],        &sAl[cur][row0 * LSTR],        LSTR, lane);
        FragPair a1 = ldA(&sAh[cur][(row0 + 16) * LSTR], &sAl[cur][(row0 + 16) * LSTR], LSTR, lane);
        FragPair b0 = ldB(&sBh[cur][col0 * LSTR],        &sBl[cur][col0 * LSTR],        LSTR, lane);
        FragPair b1 = ldB(&sBh[cur][(col0 + 16) * LSTR], &sBl[cur][(col0 + 16) * LSTR], LSTR, lane);
        acc00 = wmma3(a0, b0, acc00);
        acc01 = wmma3(a0, b1, acc01);
        acc10 = wmma3(a1, b0, acc10);
        acc11 = wmma3(a1, b1, acc11);
    }

    // ---- epilogue: C/D layout: VGPR j -> row +j (+8 lanes 16-31), col lane%16
    const int n    = lane & 15;
    const int roff = (lane & 16) ? 8 : 0;
    const float bs0 = bias ? bias[colB + col0 + n]      : 0.0f;
    const float bs1 = bias ? bias[colB + col0 + 16 + n] : 0.0f;

#pragma unroll
    for (int st = 0; st < 4; ++st) {
        const v8f acc = (st == 0) ? acc00 : (st == 1) ? acc01 : (st == 2) ? acc10 : acc11;
        const int r0  = rowB + row0 + ((st >> 1) ? 16 : 0) + roff;
        const int c   = colB + col0 + ((st & 1) ? 16 : 0) + n;
        const float bs = (st & 1) ? bs1 : bs0;
#pragma unroll
        for (int j = 0; j < 8; ++j) {
            const int r = r0 + j;
            float v = acc[j] + bs;
            if (mode == 1)      v = (v > 20.0f) ? v : log1pf(__expf(v));
            else if (mode == 2) v += skip[r * ldc + c];
            C[r * ldc + c] = v;
        }
    }
}

// ---------------------------------------------------------------------------
// Direct-global GEMM (small N=160 dbc projection), optional fused bf16 split.
// ---------------------------------------------------------------------------
__global__ void cobra_gemm_bf16x3(const __bf16* __restrict__ Ah, const __bf16* __restrict__ Al, int lda,
                                  const __bf16* __restrict__ Bh, const __bf16* __restrict__ Bl, int ldb,
                                  float* __restrict__ C, int ldc,
                                  const float* __restrict__ bias,
                                  __bf16* __restrict__ Chi, __bf16* __restrict__ Clo,
                                  int M, int N, int K) {
    const int lane = threadIdx.x & 31;
    const int wave = threadIdx.x >> 5;
    const int nn32 = N >> 5;
    const int total = (M >> 5) * nn32;
    const int t = blockIdx.x * 8 + wave;
    if (t >= total) return;

    const int row0 = (t / nn32) << 5;
    const int col0 = (t % nn32) << 5;

    v8f z = {0.f,0.f,0.f,0.f,0.f,0.f,0.f,0.f};
    v8f acc00 = z, acc01 = z, acc10 = z, acc11 = z;

    for (int k0 = 0; k0 < K; k0 += 32) {
        FragPair a0 = ldA(Ah + row0 * lda + k0,        Al + row0 * lda + k0,        lda, lane);
        FragPair a1 = ldA(Ah + (row0 + 16) * lda + k0, Al + (row0 + 16) * lda + k0, lda, lane);
        FragPair b0 = ldB(Bh + col0 * ldb + k0,        Bl + col0 * ldb + k0,        ldb, lane);
        FragPair b1 = ldB(Bh + (col0 + 16) * ldb + k0, Bl + (col0 + 16) * ldb + k0, ldb, lane);
        acc00 = wmma3(a0, b0, acc00);
        acc01 = wmma3(a0, b1, acc01);
        acc10 = wmma3(a1, b0, acc10);
        acc11 = wmma3(a1, b1, acc11);
    }

    const int n    = lane & 15;
    const int roff = (lane & 16) ? 8 : 0;
    const float bs0 = bias ? bias[col0 + n]      : 0.0f;
    const float bs1 = bias ? bias[col0 + 16 + n] : 0.0f;

#pragma unroll
    for (int st = 0; st < 4; ++st) {
        const v8f acc = (st == 0) ? acc00 : (st == 1) ? acc01 : (st == 2) ? acc10 : acc11;
        const int r0  = row0 + ((st >> 1) ? 16 : 0) + roff;
        const int c   = col0 + ((st & 1) ? 16 : 0) + n;
        const float bs = (st & 1) ? bs1 : bs0;
#pragma unroll
        for (int j = 0; j < 8; ++j) {
            const int r = r0 + j;
            float v = acc[j] + bs;
            if (C)   C[r * ldc + c] = v;
            if (Chi) split_store(v, Chi, Clo, r * ldc + c);
        }
    }
}

// ---------------------------------------------------------------------------
// Conv1d (Cin=Cout=64, k=3, pad=1) over DIM as implicit GEMM (K = 192).
// ---------------------------------------------------------------------------
__global__ void cobra_conv_silu(const float* __restrict__ xp,
                                const __bf16* __restrict__ WCh,
                                const __bf16* __restrict__ WCl,
                                const float* __restrict__ bc,
                                __bf16* __restrict__ xoneh,
                                __bf16* __restrict__ xonel) {
    const int LDW = 130;
    __shared__ float sx[64 * LDW];

    const int h0 = blockIdx.x * 128;
    const int b  = blockIdx.y;

    for (int idx = threadIdx.x; idx < 64 * LDW; idx += 256) {
        const int i = idx / LDW;
        const int tt = idx - i * LDW;
        const int h = h0 + tt - 1;
        sx[idx] = (h >= 0 && h < DIM) ? xp[(b * SEQ + i) * DIM + h] : 0.0f;
    }
    __syncthreads();

    const int lane = threadIdx.x & 31;
    const int wave = threadIdx.x >> 5;
    const int hb   = wave * 16;

    v8f z = {0.f,0.f,0.f,0.f,0.f,0.f,0.f,0.f};
    v8f acc[4] = {z, z, z, z};

    const int n  = lane & 15;
    const int kh = (lane & 16) ? 16 : 0;

    for (int k0 = 0; k0 < 192; k0 += 32) {
        FragPair bf;
#pragma unroll
        for (int j = 0; j < 16; ++j) {
            const int kk  = k0 + kh + j;
            const int i   = kk / 3;
            const int tap = kk - 3 * i;
            const float v = sx[i * LDW + hb + n + tap];
            __bf16 h = (__bf16)v;
            bf.hi[j] = h;
            bf.lo[j] = (__bf16)(v - (float)h);
        }
#pragma unroll
        for (int mt = 0; mt < 4; ++mt) {
            FragPair af = ldA(WCh + (mt * 16) * 192 + k0, WCl + (mt * 16) * 192 + k0, 192, lane);
            acc[mt] = wmma3(af, bf, acc[mt]);
        }
    }

    const int roff = (lane & 16) ? 8 : 0;
#pragma unroll
    for (int mt = 0; mt < 4; ++mt) {
#pragma unroll
        for (int j = 0; j < 8; ++j) {
            const int o = mt * 16 + roff + j;
            const float v = silu_f(acc[mt][j] + bc[o]);
            split_store(v, xoneh, xonel, (b * SEQ + o) * DIM + h0 + hb + n);
        }
    }
}

// ---------------------------------------------------------------------------
// Selective scan (L=64, N=16 state in registers) fused with +D*u and SiLU
// gate; u reconstructed from its bf16 hi/lo split; output bf16 hi/lo.
// ---------------------------------------------------------------------------
__global__ void cobra_scan_gate(const __bf16* __restrict__ uh,
                                const __bf16* __restrict__ ul,
                                const float* __restrict__ delta,
                                const float* __restrict__ dbc,
                                const float* __restrict__ A_log,
                                const float* __restrict__ Dv,
                                const float* __restrict__ gate,
                                __bf16* __restrict__ outh,
                                __bf16* __restrict__ outl) {
    __shared__ float sB[SEQ * DSTATE];
    __shared__ float sC[SEQ * DSTATE];

    const int b = blockIdx.y;
    const int e = blockIdx.x * blockDim.x + threadIdx.x;

    for (int idx = threadIdx.x; idx < SEQ * DSTATE; idx += blockDim.x) {
        const int l = idx >> 4;
        const int nn = idx & 15;
        sB[idx] = dbc[(b * SEQ + l) * 160 + DTRANK + nn];
        sC[idx] = dbc[(b * SEQ + l) * 160 + DTRANK + DSTATE + nn];
    }
    __syncthreads();

    float Arow[DSTATE];
#pragma unroll
    for (int nn = 0; nn < DSTATE; ++nn) Arow[nn] = -__expf(A_log[e * DSTATE + nn]);
    const float Dd = Dv[e];

    float h[DSTATE];
#pragma unroll
    for (int nn = 0; nn < DSTATE; ++nn) h[nn] = 0.0f;

    for (int l = 0; l < SEQ; ++l) {
        const int m = b * SEQ + l;
        const float d  = delta[m * DIM + e];
        const float u  = (float)uh[m * DIM + e] + (float)ul[m * DIM + e];
        const float du = d * u;
        float y = 0.0f;
#pragma unroll
        for (int nn = 0; nn < DSTATE; ++nn) {
            const float dA = __expf(d * Arow[nn]);
            h[nn] = dA * h[nn] + du * sB[l * DSTATE + nn];
            y += h[nn] * sC[l * DSTATE + nn];
        }
        const float g = gate[m * DIM + e];
        split_store((y + Dd * u) * silu_f(g), outh, outl, m * DIM + e);
    }
}

// ---------------------------------------------------------------------------
extern "C" void kernel_launch(void* const* d_in, const int* in_sizes, int n_in,
                              void* d_out, int out_size, void* d_ws, size_t ws_size,
                              hipStream_t stream) {
    (void)in_sizes; (void)n_in; (void)out_size; (void)ws_size;

    const float* x      = (const float*)d_in[0];
    const float* W_proj = (const float*)d_in[1];
    const float* b_proj = (const float*)d_in[2];
    const float* W_conv = (const float*)d_in[3];
    const float* b_conv = (const float*)d_in[4];
    const float* W_dbc  = (const float*)d_in[5];
    const float* W_dt   = (const float*)d_in[6];
    const float* b_dt   = (const float*)d_in[7];
    const float* A_log  = (const float*)d_in[8];
    const float* Dv     = (const float*)d_in[9];
    float* out          = (float*)d_out;

    char* ws = (char*)d_ws;
    const size_t MB = 1024 * 1024;
    float*  XP     = (float*) (ws + 0 * MB);
    __bf16* XONEh  = (__bf16*)(ws + 8 * MB);
    __bf16* XONEl  = (__bf16*)(ws + 12 * MB);
    float*  DBC    = (float*) (ws + 16 * MB);
    __bf16* DBCh   = (__bf16*)(ws + 17 * MB);
    __bf16* DBCl   = (__bf16*)(ws + 17 * MB + 512 * 1024);
    float*  DELTA  = (float*) (ws + 18 * MB);
    __bf16* GATEDh = (__bf16*)(ws + 26 * MB);
    __bf16* GATEDl = (__bf16*)(ws + 30 * MB);
    __bf16* WPh    = (__bf16*)(ws + 34 * MB);
    __bf16* WPl    = (__bf16*)(ws + 42 * MB);
    __bf16* Xh     = (__bf16*)(ws + 50 * MB);
    __bf16* Xl     = (__bf16*)(ws + 54 * MB);
    __bf16* WDBCh  = (__bf16*)(ws + 58 * MB);
    __bf16* WDBCl  = (__bf16*)(ws + 59 * MB);
    __bf16* WDTh   = (__bf16*)(ws + 60 * MB);
    __bf16* WDTl   = (__bf16*)(ws + 61 * MB);
    __bf16* WCh    = (__bf16*)(ws + 62 * MB);
    __bf16* WCl    = (__bf16*)(ws + 62 * MB + 64 * 1024);

    // --- one-time operand splits ---
    cobra_split<<<(DIM * DIM + 255) / 256, 256, 0, stream>>>(W_proj, WPh, WPl, DIM * DIM);
    cobra_split<<<(MROWS * DIM + 255) / 256, 256, 0, stream>>>(x, Xh, Xl, MROWS * DIM);
    cobra_split<<<(160 * DIM + 255) / 256, 256, 0, stream>>>(W_dbc, WDBCh, WDBCl, 160 * DIM);
    cobra_split<<<(DIM * DTRANK + 255) / 256, 256, 0, stream>>>(W_dt, WDTh, WDTl, DIM * DTRANK);
    cobra_split<<<(64 * 192 + 255) / 256, 256, 0, stream>>>(W_conv, WCh, WCl, 64 * 192);

    // 1) XP = x @ W_proj^T + b_proj            (1024,2048,K=2048) LDS-tiled
    cobra_gemm_lds<<<(MROWS / 64) * (DIM / 128), 256, 0, stream>>>(
        Xh, Xl, DIM, WPh, WPl, DIM, XP, DIM, b_proj, nullptr,
        MROWS, DIM, DIM, 0);

    // 2) XONE = silu(conv1d(XP) + b_conv)      -> bf16 hi/lo
    {
        dim3 grid(DIM / 128, BATCH);
        cobra_conv_silu<<<grid, 256, 0, stream>>>(XP, WCh, WCl, b_conv, XONEh, XONEl);
    }

    // 3) DBC = XONE @ W_dbc^T                  (1024,160,K=2048) direct + split
    cobra_gemm_bf16x3<<<(32 * 5 + 7) / 8, 256, 0, stream>>>(
        XONEh, XONEl, DIM, WDBCh, WDBCl, DIM, DBC, 160, nullptr, DBCh, DBCl,
        MROWS, 160, DIM);

    // 4) DELTA = softplus(DBC[:,:128] @ W_dt^T + b_dt)  (1024,2048,K=128) LDS-tiled
    cobra_gemm_lds<<<(MROWS / 64) * (DIM / 128), 256, 0, stream>>>(
        DBCh, DBCl, 160, WDTh, WDTl, DTRANK, DELTA, DIM, b_dt, nullptr,
        MROWS, DIM, DTRANK, 1);

    // 5) GATED = (selective_scan + D*u) * silu(XP)  -> bf16 hi/lo
    {
        dim3 grid(DIM / 256, BATCH);
        cobra_scan_gate<<<grid, 256, 0, stream>>>(
            XONEh, XONEl, DELTA, DBC, A_log, Dv, XP, GATEDh, GATEDl);
    }

    // 6) out = GATED @ W_proj^T + b_proj + x   (1024,2048,K=2048) LDS-tiled
    cobra_gemm_lds<<<(MROWS / 64) * (DIM / 128), 256, 0, stream>>>(
        GATEDh, GATEDl, DIM, WPh, WPl, DIM, out, DIM, b_proj, x,
        MROWS, DIM, DIM, 2);
}
